// GCN_12360915878594
// MI455X (gfx1250) — matrix operand order repaired and checked by reading, not compiled, not run on hardware
//
#include <hip/hip_runtime.h>

// ---------------------------------------------------------------------------
// Types for CDNA5 WMMA (wave32): v_wmma_f32_16x16x32_f16
// ---------------------------------------------------------------------------
typedef __attribute__((ext_vector_type(16))) _Float16 v16h;
typedef __attribute__((ext_vector_type(8)))  _Float16 v8h;
typedef __attribute__((ext_vector_type(8)))  float    v8f;
typedef __attribute__((ext_vector_type(4)))  unsigned int u32x4;
typedef __attribute__((ext_vector_type(4)))  int      i32x4;
typedef __attribute__((ext_vector_type(8)))  int      i32x8;

#define BATCH 16
#define TPB   256
#define BM    128
#define BN    128
#define BK    32
#define LDA   40   // LDS pitch of A tile rows (halves): 80B, 16B-aligned rows
#define LDBT  40   // LDS pitch of transposed-B tile rows (halves)

#if __has_builtin(__builtin_amdgcn_tensor_load_to_lds) && __has_builtin(__builtin_amdgcn_s_wait_tensorcnt)
#define HAVE_TDM 1
#else
#define HAVE_TDM 0
#endif

// ---------------------------------------------------------------------------
// Elementwise helpers
// ---------------------------------------------------------------------------
__global__ void f32_to_f16_kernel(const float* __restrict__ in,
                                  _Float16* __restrict__ out, long n) {
  long i = (long)blockIdx.x * blockDim.x + threadIdx.x;
  if (i < n) out[i] = (_Float16)in[i];
}

__global__ void zero_f32_kernel(float* __restrict__ p, long n) {
  long i = (long)blockIdx.x * blockDim.x + threadIdx.x;
  if (i < n) p[i] = 0.0f;
}

// ---------------------------------------------------------------------------
// TDM: DMA the A tile (rows [m0,m0+128) x K-slice [k0,k0+32) of row-major
// f16 A[M,K]) into LDS with a 40-half row pitch. TDM OOB reads return zero,
// which reproduces the zero-padding of the guarded fill.
//   pad: every 16 DWORDs (32 halves) insert 4 DWORDs (8 halves) -> pitch 40.
// Issued by one wave; completion via s_wait_tensorcnt then the block barrier.
// ---------------------------------------------------------------------------
#if HAVE_TDM
__device__ __forceinline__ void tdm_load_tile_a(const _Float16* A, void* ldsDst,
                                                int m0, int k0, int M, int K) {
  const unsigned lds = (unsigned)(unsigned long long)ldsDst;
  const unsigned long long ga =
      (unsigned long long)(const void*)(A + (long)m0 * K + k0);
  u32x4 g0;
  g0[0] = 1u;                                   // count=1, user mode
  g0[1] = lds;                                  // lds_addr (bytes)
  g0[2] = (unsigned)ga;                         // global_addr[31:0]
  g0[3] = (unsigned)((ga >> 32) & 0x01FFFFFFu)  // global_addr[56:32]
          | 0x80000000u;                        // type=2 ("image")
  const unsigned td0 = (unsigned)(K - k0);      // remaining elems along K
  const unsigned td1 = (unsigned)(M - m0);      // remaining rows
  i32x8 g1;
  g1[0] = (int)((1u << 16)        // data_size = 2 bytes
                | (1u << 20)      // pad_enable
                | (3u << 22)      // pad_interval: 16 DWORDs
                | (3u << 25));    // pad_amount: 4 DWORDs
  g1[1] = (int)((td0 & 0xFFFFu) << 16);                                  // tensor_dim0 lo
  g1[2] = (int)(((td0 >> 16) & 0xFFFFu) | ((td1 & 0xFFFFu) << 16));      // td0 hi | td1 lo
  g1[3] = (int)(((td1 >> 16) & 0xFFFFu) | ((unsigned)BK << 16));         // td1 hi | tile_dim0=32
  g1[4] = (int)BM;                              // tile_dim1=128, tile_dim2=0
  g1[5] = (int)K;                               // tensor_dim0_stride lo
  g1[6] = 0;                                    // stride hi | dim1_stride lo
  g1[7] = 0;
  const i32x4 z4 = {0, 0, 0, 0};
#if defined(__clang_major__) && __clang_major__ >= 23
  const i32x8 z8 = {0, 0, 0, 0, 0, 0, 0, 0};
  __builtin_amdgcn_tensor_load_to_lds(g0, g1, z4, z4, z8, 0);
#else
  __builtin_amdgcn_tensor_load_to_lds(g0, g1, z4, z4, 0);
#endif
  __builtin_amdgcn_s_wait_tensorcnt(0);
}
#endif

// ---------------------------------------------------------------------------
// Core wave32 WMMA micro-tile: each wave computes 32(M) x 64(N).
// LDS layouts (CDNA5 ISA 7.12.2):
//   sA  row-major  [row][k]  pitch LDA:  A frag = 2x ds_load_b128
//   sBt col-major  [col][k]  pitch LDBT: B frag = 2x ds_load_b128
// ---------------------------------------------------------------------------
__device__ __forceinline__ void wmma_wave_tile(const _Float16* __restrict__ sA,
                                               const _Float16* __restrict__ sBt,
                                               v8f (&acc)[2][4],
                                               int wm, int wn, int lane) {
  const int half = lane >> 4;
  const int lr   = lane & 15;
  v16h af[2];
  v16h bf[4];
#pragma unroll
  for (int mt = 0; mt < 2; ++mt) {
    const _Float16* ap = sA + (wm * 32 + mt * 16 + lr) * LDA;
    v8h lo = *(const v8h*)(ap + 8 * half);        // K = 8*half .. +7
    v8h hi = *(const v8h*)(ap + 16 + 8 * half);   // K = 16+8*half .. +7
    af[mt] = __builtin_shufflevector(lo, hi, 0, 1, 2, 3, 4, 5, 6, 7,
                                     8, 9, 10, 11, 12, 13, 14, 15);
  }
#pragma unroll
  for (int nt = 0; nt < 4; ++nt) {
    const _Float16* bp = sBt + (wn * 64 + nt * 16 + lr) * LDBT;
    v8h lo = *(const v8h*)(bp + 8 * half);        // K = 8*half .. +7
    v8h hi = *(const v8h*)(bp + 16 + 8 * half);   // K = 16+8*half .. +7
    bf[nt] = __builtin_shufflevector(lo, hi, 0, 1, 2, 3, 4, 5, 6, 7,
                                     8, 9, 10, 11, 12, 13, 14, 15);
  }
#pragma unroll
  for (int mt = 0; mt < 2; ++mt)
#pragma unroll
    for (int nt = 0; nt < 4; ++nt)
      acc[mt][nt] = __builtin_amdgcn_wmma_f32_16x16x32_f16(
          false, af[mt], false, bf[nt], (short)0, acc[mt][nt], false, false);
}

// ---------------------------------------------------------------------------
// Generic GEMM: C[M,N] (f32) = A[M,K] (f16, row-major) * B[K,N] (f16, row-major)
// ---------------------------------------------------------------------------
__global__ void __launch_bounds__(TPB)
gemm_f16_kernel(const _Float16* __restrict__ A, const _Float16* __restrict__ B,
                float* __restrict__ C, int M, int N, int K) {
  __shared__ _Float16 sA[BM * LDA];
  __shared__ _Float16 sBt[BN * LDBT];
  const int tid  = threadIdx.x;
  const int wave = tid >> 5;
  const int lane = tid & 31;
  const int wm = wave >> 1;
  const int wn = wave & 1;
  const int m0 = blockIdx.y * BM;
  const int n0 = blockIdx.x * BN;
  const bool vecA = ((K & 7) == 0);
  const bool vecB = ((N & 7) == 0);

  v8f acc[2][4];
  v8f zv = {};
#pragma unroll
  for (int a = 0; a < 2; ++a)
#pragma unroll
    for (int b = 0; b < 4; ++b) acc[a][b] = zv;

  for (int k0 = 0; k0 < K; k0 += BK) {
    // ---- A tile -> sA (row-major, pitch LDA) ----
#if HAVE_TDM
    if (tid < 32) tdm_load_tile_a(A, (void*)sA, m0, k0, M, K);
#else
    if (vecA) {
      for (int i = tid; i < BM * (BK / 8); i += TPB) {
        const int r = i >> 2, g = i & 3;
        const int gm = m0 + r, gk = k0 + g * 8;
        v8h v = {};
        if (gm < M && gk + 8 <= K) v = *(const v8h*)(A + (long)gm * K + gk);
        *(v8h*)(sA + r * LDA + g * 8) = v;
      }
    } else {
      for (int i = tid; i < BM * BK; i += TPB) {
        const int r = i / BK, c = i % BK;
        const int gm = m0 + r, gk = k0 + c;
        _Float16 v = (_Float16)0.0f;
        if (gm < M && gk < K) v = A[(long)gm * K + gk];
        sA[r * LDA + c] = v;
      }
    }
#endif
    // ---- B tile -> sBt (transposed: [n][k], pitch LDBT) ----
    if (vecB) {
      for (int i = tid; i < BK * (BN / 8); i += TPB) {
        const int r = i / (BN / 8);        // k row within tile
        const int g = i % (BN / 8);        // col group
        const int gk = k0 + r;
        const int gn = n0 + g * 8;
        v8h v = {};
        if (gk < K && gn < N) v = *(const v8h*)(B + (long)gk * N + gn);
#pragma unroll
        for (int j = 0; j < 8; ++j) sBt[(g * 8 + j) * LDBT + r] = v[j];
      }
    } else {
      for (int i = tid; i < BK * BN; i += TPB) {
        const int r = i / BN, c = i % BN;
        const int gk = k0 + r, gn = n0 + c;
        _Float16 v = (_Float16)0.0f;
        if (gk < K && gn < N) v = B[(long)gk * N + gn];
        sBt[c * LDBT + r] = v;
      }
    }
    // prefetch next B tile rows into GL2 (global_prefetch_b8)
    if (k0 + BK < K) {
      const int pr = k0 + BK + (tid & 31);
      if (pr < K) __builtin_prefetch(B + (long)pr * N + n0 + ((tid >> 5) << 4), 0, 1);
    }
    __syncthreads();
    wmma_wave_tile(sA, sBt, acc, wm, wn, lane);
    __syncthreads();
  }

  const int half = lane >> 4;
  const int lr   = lane & 15;
#pragma unroll
  for (int nt = 0; nt < 4; ++nt) {
    const int n = n0 + wn * 64 + nt * 16 + lr;
    if (n >= N) continue;
#pragma unroll
    for (int mt = 0; mt < 2; ++mt)
#pragma unroll
      for (int vv = 0; vv < 8; ++vv) {
        const int m = m0 + wm * 32 + mt * 16 + vv + 8 * half;
        if (m < M) C[(long)m * N + n] = acc[mt][nt][vv];
      }
  }
}

// ---------------------------------------------------------------------------
// Implicit-GEMM convolution (NCHW, B=16):
//   Y = relu(W (x) X + bias) (+ residual).  im2col fused into B-tile fill.
// ---------------------------------------------------------------------------
__global__ void __launch_bounds__(TPB)
conv_gemm_f16_kernel(const _Float16* __restrict__ Wf, const float* __restrict__ X,
                     const float* __restrict__ bias, const float* __restrict__ res,
                     float* __restrict__ Y,
                     int O, int Ci, int Hi, int Wi,
                     int kz, int stride, int padT, int padL, int Ho, int Wo) {
  __shared__ _Float16 sA[BM * LDA];
  __shared__ _Float16 sBt[BN * LDBT];
  const int K  = Ci * kz * kz;
  const int HW = Ho * Wo;
  const int N  = BATCH * HW;
  const int kz2 = kz * kz;

  const int tid  = threadIdx.x;
  const int wave = tid >> 5;
  const int lane = tid & 31;
  const int wm = wave >> 1;
  const int wn = wave & 1;
  const int m0 = blockIdx.y * BM;
  const int n0 = blockIdx.x * BN;

  v8f acc[2][4];
  v8f zv = {};
#pragma unroll
  for (int a = 0; a < 2; ++a)
#pragma unroll
    for (int b = 0; b < 4; ++b) acc[a][b] = zv;

  for (int k0 = 0; k0 < K; k0 += BK) {
    // ---- A tile: pre-converted f16 weights [O, K] ----
#if HAVE_TDM
    if (tid < 32) tdm_load_tile_a(Wf, (void*)sA, m0, k0, O, K);
#else
    for (int i = tid; i < BM * BK; i += TPB) {
      const int r = i / BK, c = i % BK;
      const int gm = m0 + r, gk = k0 + c;
      _Float16 v = (_Float16)0.0f;
      if (gm < O && gk < K) v = Wf[(long)gm * K + gk];
      sA[r * LDA + c] = v;
    }
#endif
    // ---- B tile: on-the-fly im2col gather, stored transposed ----
    for (int i = tid; i < BK * BN; i += TPB) {
      const int r = i / BN, c = i % BN;
      const int gk = k0 + r, gn = n0 + c;
      _Float16 v = (_Float16)0.0f;
      if (gk < K && gn < N) {
        const int ci = gk / kz2;
        const int rr = gk % kz2;
        const int kh = rr / kz, kw = rr % kz;
        const int b  = gn / HW;
        const int s  = gn % HW;
        const int ho = s / Wo, wo = s % Wo;
        const int h = ho * stride - padT + kh;
        const int w = wo * stride - padL + kw;
        if (h >= 0 && h < Hi && w >= 0 && w < Wi)
          v = (_Float16)X[(((long)b * Ci + ci) * Hi + h) * Wi + w];
      }
      sBt[c * LDBT + r] = v;
    }
    __syncthreads();
    wmma_wave_tile(sA, sBt, acc, wm, wn, lane);
    __syncthreads();
  }

  const int half = lane >> 4;
  const int lr   = lane & 15;
#pragma unroll
  for (int nt = 0; nt < 4; ++nt) {
    const int n = n0 + wn * 64 + nt * 16 + lr;
    if (n >= N) continue;
    const int b = n / HW;
    const int s = n % HW;
    const long base = (long)b * O * HW + s;
#pragma unroll
    for (int mt = 0; mt < 2; ++mt)
#pragma unroll
      for (int vv = 0; vv < 8; ++vv) {
        const int m = m0 + wm * 32 + mt * 16 + vv + 8 * half;
        if (m < O) {
          float y = acc[mt][nt][vv] + bias[m];
          y = fmaxf(y, 0.0f);
          const long oi = base + (long)m * HW;
          if (res) y += res[oi];
          Y[oi] = y;
        }
      }
  }
}

// ---------------------------------------------------------------------------
// Perceptual-feature projection: bilinear sample of 4 feature pyramids.
// ---------------------------------------------------------------------------
__device__ __forceinline__ float bilin_sample(const float* __restrict__ f, int s,
                                              float h, float w) {
  const float sc = (float)s / 224.0f;
  const float smax = (float)(s - 1);
  float x = fminf(fmaxf(h * sc, 0.0f), smax);
  float y = fminf(fmaxf(w * sc, 0.0f), smax);
  const float x1 = floorf(x), y1 = floorf(y);
  const float x2 = fminf(x1 + 1.0f, smax), y2 = fminf(y1 + 1.0f, smax);
  const int xi1 = (int)x1, yi1 = (int)y1, xi2 = (int)x2, yi2 = (int)y2;
  const float wx = x - x1, wy = y - y1;
  const float q11 = f[xi1 * s + yi1];
  const float q21 = f[xi2 * s + yi1];
  const float q12 = f[xi1 * s + yi2];
  const float q22 = f[xi2 * s + yi2];
  return q11 * (1.0f - wx) * (1.0f - wy) + q21 * wx * (1.0f - wy) +
         q12 * (1.0f - wx) * wy + q22 * wx * wy;
}

__global__ void proj_kernel(const float* __restrict__ coords, int n,
                            const float* __restrict__ f2, const float* __restrict__ f3,
                            const float* __restrict__ f4, const float* __restrict__ f5,
                            float* __restrict__ out, int ostride) {
  const int b = blockIdx.x / n;
  const int v = blockIdx.x % n;
  const float* c = coords + ((long)b * n + v) * 3;
  const float X = c[0], Y = c[1], Z = c[2];
  float d = -Z;
  d = (fabsf(d) < 0.001f) ? 0.001f : d;
  const float h = fminf(fmaxf(250.0f * (-Y) / d + 112.0f, 0.0f), 223.0f);
  const float w = fminf(fmaxf(250.0f * X / d + 112.0f, 0.0f), 223.0f);
  float* orow = out + ((long)b * n + v) * ostride;
  const float* b2 = f2 + (long)b * 64 * 56 * 56;
  const float* b3 = f3 + (long)b * 128 * 28 * 28;
  const float* b4 = f4 + (long)b * 256 * 14 * 14;
  const float* b5 = f5 + (long)b * 512 * 7 * 7;
  for (int cc = threadIdx.x; cc < 963; cc += blockDim.x) {
    float val;
    if (cc < 3)        val = c[cc];
    else if (cc < 67)  val = bilin_sample(b2 + (long)(cc - 3)   * 56 * 56, 56, h, w);
    else if (cc < 195) val = bilin_sample(b3 + (long)(cc - 67)  * 28 * 28, 28, h, w);
    else if (cc < 451) val = bilin_sample(b4 + (long)(cc - 195) * 14 * 14, 14, h, w);
    else               val = bilin_sample(b5 + (long)(cc - 451) * 7 * 7,   7,  h, w);
    orow[cc] = val;
  }
}

// ---------------------------------------------------------------------------
// SPMM (segment-sum over random rows) via fp32 atomics; batch in registers.
// ---------------------------------------------------------------------------
__global__ void spmm_kernel(const int* __restrict__ rows, const int* __restrict__ cols,
                            const float* __restrict__ vals, const float* __restrict__ xw,
                            float* __restrict__ out, int E, int n, int co) {
  long i = (long)blockIdx.x * blockDim.x + threadIdx.x;
  if (i >= (long)E * co) return;
  const int e = (int)(i / co);
  const int c = (int)(i % co);
  const float v = vals[e];
  const int r  = rows[e];
  const int cl = cols[e];
#pragma unroll
  for (int b = 0; b < BATCH; ++b)
    atomicAdd(&out[((long)b * n + r) * co + c], v * xw[((long)b * n + cl) * co + c]);
}

// y = h0*d0[node] + spmm + bias[c]; optional relu; optional (y+prev2)*0.5
__global__ void gcn_combine_kernel(const float* __restrict__ h0, const float* __restrict__ sp,
                                   const float* __restrict__ d0, const float* __restrict__ bias,
                                   const float* __restrict__ prev2, float* __restrict__ out,
                                   int n, int co, int doRelu, long total) {
  long i = (long)blockIdx.x * blockDim.x + threadIdx.x;
  if (i >= total) return;
  const int c = (int)(i % co);
  const long r = i / co;
  const int node = (int)(r % n);
  float y = h0[i] * d0[node] + sp[i] + bias[c];
  if (doRelu) y = fmaxf(y, 0.0f);
  if (prev2) y = 0.5f * (y + prev2[i]);
  out[i] = y;
}

// Graph unpool: rows [0,n) copied; rows [n,n+extra) = mean of two parents.
__global__ void unpool_kernel(const float* __restrict__ src, const int* __restrict__ pidx,
                              float* __restrict__ dst, int n, int extra, int C) {
  const long total = (long)BATCH * (n + extra) * C;
  long i = (long)blockIdx.x * blockDim.x + threadIdx.x;
  if (i >= total) return;
  const int c = (int)(i % C);
  const long r = i / C;
  const int row = (int)(r % (n + extra));
  const int b = (int)(r / (n + extra));
  float v;
  if (row < n) {
    v = src[((long)b * n + row) * C + c];
  } else {
    const int j = row - n;
    const int a0 = pidx[2 * j], a1 = pidx[2 * j + 1];
    v = 0.5f * (src[((long)b * n + a0) * C + c] + src[((long)b * n + a1) * C + c]);
  }
  dst[i] = v;
}

// Copy (B,n,C) contiguous -> rows of stride dstStride at column offset dstOff.
__global__ void concat_copy_kernel(const float* __restrict__ src, float* __restrict__ dst,
                                   int n, int C, int dstStride, int dstOff) {
  const long total = (long)BATCH * n * C;
  long i = (long)blockIdx.x * blockDim.x + threadIdx.x;
  if (i >= total) return;
  const int c = (int)(i % C);
  const long r = i / C;
  dst[r * dstStride + dstOff + c] = src[i];
}

// ---------------------------------------------------------------------------
// Host orchestration
// ---------------------------------------------------------------------------
struct SupT { const int* rows; const int* cols; const float* vals; const float* d0; int n; int E; };

extern "C" void kernel_launch(void* const* d_in, const int* in_sizes, int n_in,
                              void* d_out, int out_size, void* d_ws, size_t ws_size,
                              hipStream_t stream) {
  (void)in_sizes; (void)n_in; (void)out_size; (void)ws_size;

  // ---- input unpack (setup_inputs dict insertion order) ----
  const float* img      = (const float*)d_in[0];
  const float* features = (const float*)d_in[1];
  const float* convW[18]; const float* convBs[18];
  for (int l = 0; l < 18; ++l) { convW[l] = (const float*)d_in[2 + 2 * l]; convBs[l] = (const float*)d_in[3 + 2 * l]; }
  const float *gW0[45], *gW1[45], *gB[45];
  for (int g = 0; g < 45; ++g) {
    gW0[g] = (const float*)d_in[38 + 3 * g];
    gW1[g] = (const float*)d_in[39 + 3 * g];
    gB[g]  = (const float*)d_in[40 + 3 * g];
  }
  SupT S1 = { (const int*)d_in[173], (const int*)d_in[174], (const float*)d_in[175], (const float*)d_in[176], 156, 936 };
  SupT S2 = { (const int*)d_in[177], (const int*)d_in[178], (const float*)d_in[179], (const float*)d_in[180], 618, 3708 };
  SupT S3 = { (const int*)d_in[181], (const int*)d_in[182], (const float*)d_in[183], (const float*)d_in[184], 2466, 14796 };
  const int* pool1 = (const int*)d_in[185];
  const int* pool2 = (const int*)d_in[186];
  float* outF = (float*)d_out;

  // ---- bump allocator over workspace ----
  char* wsBase = (char*)d_ws;
  size_t wsOff = 0;
  auto alloc = [&](size_t bytes) -> void* {
    wsOff = (wsOff + 255) & ~(size_t)255;
    void* p = wsBase + wsOff;
    wsOff += bytes;
    return p;
  };
  auto gs = [](long tot) { return dim3((unsigned)((tot + TPB - 1) / TPB)); };
  auto convert = [&](const float* src, _Float16* dst, long n) {
    f32_to_f16_kernel<<<gs(n), TPB, 0, stream>>>(src, dst, n);
  };

  // =========================================================================
  // Phase A: ResNet (implicit-GEMM WMMA convs)
  // spec: O, I, k, stride, padT, padL, Hin, Hout, resFrom
  // =========================================================================
  static const int spec[18][9] = {
    { 16,   3, 3, 1, 1, 1, 224, 224, -1},
    { 16,  16, 3, 1, 1, 1, 224, 224, -1},
    { 32,  16, 3, 2, 0, 0, 224, 112, -1},
    { 32,  32, 3, 1, 1, 1, 112, 112, -1},
    { 32,  32, 3, 1, 1, 1, 112, 112,  2},
    { 64,  32, 3, 2, 0, 0, 112,  56, -1},
    { 64,  64, 3, 1, 1, 1,  56,  56, -1},
    { 64,  64, 3, 1, 1, 1,  56,  56,  5},
    {128,  64, 3, 2, 0, 0,  56,  28, -1},
    {128, 128, 3, 1, 1, 1,  28,  28, -1},
    {128, 128, 3, 1, 1, 1,  28,  28,  8},
    {256, 128, 5, 2, 1, 1,  28,  14, -1},
    {256, 256, 3, 1, 1, 1,  14,  14, -1},
    {256, 256, 3, 1, 1, 1,  14,  14, 11},
    {512, 256, 5, 2, 1, 1,  14,   7, -1},
    {512, 512, 3, 1, 1, 1,   7,   7, -1},
    {512, 512, 3, 1, 1, 1,   7,   7, 14},
    {512, 512, 3, 1, 1, 1,   7,   7, -1},
  };

  float* act[18];
  const float* curin = img;
  for (int l = 0; l < 18; ++l) {
    const int O = spec[l][0], I = spec[l][1], k = spec[l][2], st = spec[l][3];
    const int pT = spec[l][4], pL = spec[l][5], Hi = spec[l][6], Ho = spec[l][7];
    const int rf = spec[l][8];
    const long wn = (long)O * I * k * k;
    _Float16* wh = (_Float16*)alloc((size_t)wn * sizeof(_Float16));
    convert(convW[l], wh, wn);
    act[l] = (float*)alloc((size_t)BATCH * O * Ho * Ho * sizeof(float));
    const int N = BATCH * Ho * Ho;
    dim3 grid((unsigned)((N + BN - 1) / BN), (unsigned)((O + BM - 1) / BM));
    const float* res = (rf >= 0) ? act[rf] : nullptr;
    conv_gemm_f16_kernel<<<grid, TPB, 0, stream>>>(wh, curin, convBs[l], res, act[l],
                                                   O, I, Hi, Hi, k, st, pT, pL, Ho, Ho);
    curin = act[l];
  }
  const float* x2 = act[7];   // 64  @ 56x56
  const float* x3 = act[10];  // 128 @ 28x28
  const float* x4 = act[13];  // 256 @ 14x14
  const float* x5 = act[17];  // 512 @ 7x7

  // =========================================================================
  // Phase B: GCN blocks with projection / unpool
  // =========================================================================
  const size_t actMax = (size_t)BATCH * 2466 * 1155;
  float* G[3];
  for (int i = 0; i < 3; ++i) G[i] = (float*)alloc(actMax * sizeof(float));
  _Float16* xh  = (_Float16*)alloc(actMax * sizeof(_Float16));
  _Float16* w0h = (_Float16*)alloc((size_t)1155 * 192 * sizeof(_Float16));
  _Float16* w1h = (_Float16*)alloc((size_t)1155 * 192 * sizeof(_Float16));
  const size_t hMax = (size_t)BATCH * 2466 * 192;
  float* h0buf = (float*)alloc(hMax * sizeof(float));
  float* xwbuf = (float*)alloc(hMax * sizeof(float));
  float* spbuf = (float*)alloc(hMax * sizeof(float));

  int ci_ = 0, pi_ = -1;
  auto nextBuf = [&]() { for (int i = 0; i < 3; ++i) if (i != ci_ && i != pi_) return i; return 0; };

  auto gemm = [&](const _Float16* A, const _Float16* B, float* C, int M, int N, int K) {
    dim3 grid((unsigned)((N + BN - 1) / BN), (unsigned)((M + BM - 1) / BM));
    gemm_f16_kernel<<<grid, TPB, 0, stream>>>(A, B, C, M, N, K);
  };

  auto gcnStep = [&](int g, int nNodes, int ciD, int coD, bool actF, bool eltF, const SupT& sp) {
    const int nb = nextBuf();
    const long M = (long)BATCH * nNodes;
    convert(G[ci_], xh, M * ciD);
    convert(gW0[g], w0h, (long)ciD * coD);
    convert(gW1[g], w1h, (long)ciD * coD);
    gemm(xh, w0h, h0buf, (int)M, coD, ciD);
    gemm(xh, w1h, xwbuf, (int)M, coD, ciD);
    const long tot = M * coD;
    zero_f32_kernel<<<gs(tot), TPB, 0, stream>>>(spbuf, tot);
    spmm_kernel<<<gs((long)sp.E * coD), TPB, 0, stream>>>(sp.rows, sp.cols, sp.vals,
                                                          xwbuf, spbuf, sp.E, nNodes, coD);
    const float* prev2 = eltF ? G[pi_] : nullptr;
    gcn_combine_kernel<<<gs(tot), TPB, 0, stream>>>(h0buf, spbuf, sp.d0, gB[g], prev2,
                                                    G[nb], nNodes, coD, actF ? 1 : 0, tot);
    pi_ = ci_; ci_ = nb;
  };

  auto projConcat = [&](int nNodes) {  // proj(cur) ++ prev(192) -> new (stride 1155)
    const int nb = nextBuf();
    proj_kernel<<<dim3((unsigned)(BATCH * nNodes)), TPB, 0, stream>>>(
        G[ci_], nNodes, x2, x3, x4, x5, G[nb], 1155);
    concat_copy_kernel<<<gs((long)BATCH * nNodes * 192), TPB, 0, stream>>>(
        G[pi_], G[nb], nNodes, 192, 1155, 963);
    pi_ = ci_; ci_ = nb;
  };

  auto poolStep = [&](int nNodes, int extra, const int* pidx) {
    const int nb = nextBuf();
    unpool_kernel<<<gs((long)BATCH * (nNodes + extra) * 1155), TPB, 0, stream>>>(
        G[ci_], pidx, G[nb], nNodes, extra, 1155);
    pi_ = ci_; ci_ = nb;
  };

  // ---- plan idx 0: proj(features) -> (B,156,963) ----
  proj_kernel<<<dim3((unsigned)(BATCH * 156)), TPB, 0, stream>>>(
      features, 156, x2, x3, x4, x5, G[0], 963);
  ci_ = 0; pi_ = -1;

  // ---- block 1: gc0..gc13 on 156 nodes ----
  for (int k = 0; k < 14; ++k) {
    const bool elt = (k >= 2 && k <= 12 && (k % 2 == 0));
    gcnStep(k, 156, (k == 0) ? 963 : 192, (k == 13) ? 3 : 192, k < 13, elt, S1);
  }
  // o1 = cur (B,156,3); up1 = unpool(o1, pool1)
  hipMemcpyAsync(outF, G[ci_], (size_t)BATCH * 156 * 3 * sizeof(float),
                 hipMemcpyDeviceToDevice, stream);
  unpool_kernel<<<gs((long)BATCH * 618 * 3), TPB, 0, stream>>>(
      G[ci_], pool1, outF + 7488, 156, 462, 3);

  // ---- plan idx 15/16: proj+concat, unpool -> 618 nodes ----
  projConcat(156);
  poolStep(156, 462, pool1);

  // ---- block 2: gc14..gc27 on 618 nodes ----
  for (int k = 0; k < 14; ++k) {
    const bool elt = (k >= 2 && k <= 12 && (k % 2 == 0));
    gcnStep(14 + k, 618, (k == 0) ? 1155 : 192, (k == 13) ? 3 : 192, k < 13, elt, S2);
  }
  // o2 = cur (B,618,3); up2 = unpool(o2, pool2)
  hipMemcpyAsync(outF + 37152, G[ci_], (size_t)BATCH * 618 * 3 * sizeof(float),
                 hipMemcpyDeviceToDevice, stream);
  unpool_kernel<<<gs((long)BATCH * 2466 * 3), TPB, 0, stream>>>(
      G[ci_], pool2, outF + 66816, 618, 1848, 3);

  // ---- plan idx 31/32: proj+concat, unpool -> 2466 nodes ----
  projConcat(618);
  poolStep(618, 1848, pool2);

  // ---- block 3: gc28..gc42 on 2466 nodes ----
  for (int k = 0; k < 15; ++k) {
    const bool elt = (k >= 2 && k <= 12 && (k % 2 == 0));
    int ciD = (k == 0) ? 1155 : ((k == 14) ? 96 : 192);
    int coD = (k == 13) ? 96 : ((k == 14) ? 3 : 192);
    gcnStep(28 + k, 2466, ciD, coD, k < 14, elt, S3);
  }
  // final coords (B,2466,3)
  hipMemcpyAsync(outF + 185184, G[ci_], (size_t)BATCH * 2466 * 3 * sizeof(float),
                 hipMemcpyDeviceToDevice, stream);
}
